// GATv2_block_14388140442034
// MI455X (gfx1250) — compile-verified, matrix operand order checked by reading, not run on hardware
//
#include <hip/hip_runtime.h>
#include <hip/hip_bf16.h>

#define Nn 50000
#define Ee 800000
#define Bb 64
// NODE_IN=128, EDGE_IN=64, GLOBAL_IN=64, NODE_OUT=128, EDGE_OUT=128, GLOBAL_OUT=64

typedef __attribute__((ext_vector_type(16))) _Float16 v16h;
typedef __attribute__((ext_vector_type(8)))  float    v8f;

// ---------- WMMA fragment helpers (layouts per CDNA5 ISA 7.12.2) ----------
// A 16x32 f16: lane holds row M=lane%16; element j -> K = (lane>=16?8:0) + (j<8 ? j : j+8)
__device__ __forceinline__ v16h frag_a_f32(const float* __restrict__ row, int kbase, int half) {
  v16h a; int off = kbase + half * 8;
#pragma unroll
  for (int j = 0; j < 8; ++j) a[j]     = (_Float16)row[off + j];
#pragma unroll
  for (int j = 0; j < 8; ++j) a[j + 8] = (_Float16)row[off + 16 + j];
  return a;
}
__device__ __forceinline__ v16h frag_a_f16(const _Float16* __restrict__ row, int kbase, int half) {
  v16h a; int off = kbase + half * 8;
#pragma unroll
  for (int j = 0; j < 8; ++j) a[j]     = row[off + j];
#pragma unroll
  for (int j = 0; j < 8; ++j) a[j + 8] = row[off + 16 + j];
  return a;
}
// B 32x16 f16 from row-major f32 W[K][ncols]: lane holds col N, same K pattern as A.
__device__ __forceinline__ v16h frag_b_f32(const float* __restrict__ W, int ncols,
                                           int c, int nt, int lane) {
  int half = lane >> 4, n = nt * 16 + (lane & 15);
  v16h b;
#pragma unroll
  for (int j = 0; j < 16; ++j) {
    int k = c * 32 + half * 8 + (j < 8 ? j : j + 8);
    b[j] = (_Float16)W[(size_t)k * ncols + n];
  }
  return b;
}
__device__ __forceinline__ v8f wmma16(v16h a, v16h b, v8f c) {
  return __builtin_amdgcn_wmma_f32_16x16x32_f16(false, a, false, b, (short)0, c, false, false);
}
// Cooperative fill of B fragments into LDS: index = (c*ntiles + nt)*32 + lane
__device__ __forceinline__ void fill_b(v16h* dst, const float* W, int ncols,
                                       int nchunks, int ntiles, int tid, int nthreads) {
  int total = nchunks * ntiles * 32;
  for (int idx = tid; idx < total; idx += nthreads) {
    int lane = idx & 31, t = idx >> 5;
    int nt = t % ntiles, c = t / ntiles;
    dst[idx] = frag_b_f32(W, ncols, c, nt, lane);
  }
}
__device__ __forceinline__ void atomicMaxF(float* addr, float val) {
  unsigned int* ua = (unsigned int*)addr;
  unsigned int old = __float_as_uint(*addr);
  while (__uint_as_float(old) < val) {
    unsigned int assumed = old;
    old = atomicCAS(ua, assumed, __float_as_uint(val));
    if (old == assumed) break;
  }
}
// Blocks LICM from hoisting (and then spilling) LDS fragment loads out of the
// persistent tile loop -- keeps ds_load_b128 adjacent to each v_wmma.
#define NO_HOIST() asm volatile("" ::: "memory")

// ---------- init workspace ----------
__global__ void k_init(float* smax, float* denom, float* S, float* gsum, float* cnt) {
  int t = blockIdx.x * blockDim.x + threadIdx.x;
  if (t < Nn) { smax[t] = -3.0e38f; denom[t] = 0.f; S[t] = 0.f; }
  if (t < Bb * 128) gsum[t] = 0.f;
  if (t < Bb) cnt[t] = 0.f;
}

// ---------- edge MLP: e_new = relu([x[L] || edge_attr] @ W_e + b_e) ----------
// Persistent: weight fragments filled once per block, waves loop over tiles.
__global__ void k_edge1(const float* __restrict__ x, const int* __restrict__ ei,
                        const float* __restrict__ ea, const float* __restrict__ W_e,
                        const float* __restrict__ b_e, float* __restrict__ out_e) {
  extern __shared__ char smem[];
  v16h* Bf = (v16h*)smem;                     // 6 chunks * 8 ntiles * 32 lanes
  fill_b(Bf, W_e, 128, 6, 8, threadIdx.x, blockDim.x);
  __syncthreads();
  int wavesTotal = gridDim.x * (blockDim.x >> 5);
  int wave0 = blockIdx.x * (blockDim.x >> 5) + (threadIdx.x >> 5);
  int lane = threadIdx.x & 31, half = lane >> 4, ml = lane & 15;
  for (int tile = wave0; tile < Ee / 16; tile += wavesTotal) {
    NO_HOIST();
    int eidA = tile * 16 + ml;
    const float* xr = x + (size_t)ei[eidA] * 128;
    v16h A[6];
#pragma unroll
    for (int c = 0; c < 4; ++c) A[c] = frag_a_f32(xr, c * 32, half);
    const float* er = ea + (size_t)eidA * 64;
    A[4] = frag_a_f32(er, 0, half);
    A[5] = frag_a_f32(er, 32, half);
#pragma unroll
    for (int nt = 0; nt < 8; ++nt) {
      v8f acc = {};
#pragma unroll
      for (int c = 0; c < 6; ++c) acc = wmma16(A[c], Bf[(c * 8 + nt) * 32 + lane], acc);
      int n = nt * 16 + ml;
      float bias = b_e[n];
#pragma unroll
      for (int i = 0; i < 8; ++i) {
        float v = acc[i] + bias; v = v > 0.f ? v : 0.f;
        out_e[(size_t)(tile * 16 + half * 8 + i) * 128 + n] = v;
      }
    }
  }
}

// ---------- attention scores: leaky_relu([x[L]||x[R]||e_new]@W_le + b_le) . a ----------
__global__ void k_edge2(const float* __restrict__ x, const int* __restrict__ ei,
                        const float* __restrict__ out_e, const float* __restrict__ W_le,
                        const float* __restrict__ b_le, const float* __restrict__ avec,
                        float* __restrict__ score, float* __restrict__ smax) {
  extern __shared__ char smem[];
  v16h* Bf = (v16h*)smem;                     // 12 * 8 * 32
  fill_b(Bf, W_le, 128, 12, 8, threadIdx.x, blockDim.x);
  __syncthreads();
  int wavesTotal = gridDim.x * (blockDim.x >> 5);
  int wave0 = blockIdx.x * (blockDim.x >> 5) + (threadIdx.x >> 5);
  int lane = threadIdx.x & 31, half = lane >> 4, ml = lane & 15;
  for (int tile = wave0; tile < Ee / 16; tile += wavesTotal) {
    NO_HOIST();
    int eidA = tile * 16 + ml;
    const float* xl = x + (size_t)ei[eidA] * 128;
    const float* xr = x + (size_t)ei[Ee + eidA] * 128;
    const float* er = out_e + (size_t)eidA * 128;
    v16h A[12];
#pragma unroll
    for (int c = 0; c < 4; ++c) {
      A[c]     = frag_a_f32(xl, c * 32, half);
      A[4 + c] = frag_a_f32(xr, c * 32, half);
      A[8 + c] = frag_a_f32(er, c * 32, half);
    }
    float s[8] = {};
#pragma unroll
    for (int nt = 0; nt < 8; ++nt) {
      v8f acc = {};
#pragma unroll
      for (int c = 0; c < 12; ++c) acc = wmma16(A[c], Bf[(c * 8 + nt) * 32 + lane], acc);
      int n = nt * 16 + ml;
      float bias = b_le[n], av = avec[n];
#pragma unroll
      for (int i = 0; i < 8; ++i) {
        float v = acc[i] + bias;
        v = v > 0.f ? v : 0.2f * v;      // leaky_relu(0.2)
        s[i] += v * av;
      }
    }
#pragma unroll
    for (int i = 0; i < 8; ++i) {        // reduce over N within each half-wave (16 lanes)
      float v = s[i];
#pragma unroll
      for (int off = 8; off > 0; off >>= 1) v += __shfl_xor(v, off, 16);
      s[i] = v;
    }
    if (ml == 0) {
#pragma unroll
      for (int i = 0; i < 8; ++i) {
        int row = tile * 16 + half * 8 + i;
        score[row] = s[i];
        atomicMaxF(&smax[ei[Ee + row]], s[i]);
      }
    }
  }
}

// ---------- segment softmax scalars ----------
__global__ void k_exp(const float* __restrict__ score, const int* __restrict__ ei,
                      const float* __restrict__ smax, float* __restrict__ ex,
                      float* __restrict__ denom) {
  int e = blockIdx.x * blockDim.x + threadIdx.x;
  if (e >= Ee) return;
  int r = ei[Ee + e];
  float v = __expf(score[e] - smax[r]);
  ex[e] = v;
  atomicAdd(&denom[r], v);
}
__global__ void k_alpha(const int* __restrict__ ei, const float* __restrict__ ex,
                        const float* __restrict__ denom, float* __restrict__ S) {
  int e = blockIdx.x * blockDim.x + threadIdx.x;
  if (e >= Ee) return;
  int r = ei[Ee + e];
  atomicAdd(&S[r], ex[e] / (denom[r] + 1e-16f));
}
__global__ void k_cnt(const int* __restrict__ batch, float* __restrict__ cnt) {
  int i = blockIdx.x * blockDim.x + threadIdx.x;
  if (i >= Nn) return;
  atomicAdd(&cnt[batch[i]], 1.f);
}

// ---------- node model: agg = (x@W_n + b_n)*S ; x_new = relu([agg||glob[batch]]@W_n2 + b_n2)
// Persistent with block-uniform base so the staging __syncthreads stays uniform.
__global__ void k_node(const float* __restrict__ x, const int* __restrict__ batch,
                       const float* __restrict__ glob, const float* __restrict__ W_n,
                       const float* __restrict__ b_n, const float* __restrict__ W_n2,
                       const float* __restrict__ b_n2, const float* __restrict__ S,
                       float* __restrict__ out_x, float* __restrict__ gsum) {
  extern __shared__ char smem[];
  v16h* Bn  = (v16h*)smem;                    // 4*8*32
  v16h* Bn2 = Bn + 4 * 8 * 32;                // 6*8*32
  _Float16* stage = (_Float16*)(Bn2 + 6 * 8 * 32);   // nwaves * 16 * 128
  fill_b(Bn,  W_n,  128, 4, 8, threadIdx.x, blockDim.x);
  fill_b(Bn2, W_n2, 128, 6, 8, threadIdx.x, blockDim.x);
  __syncthreads();
  int wpb = blockDim.x >> 5;
  int wib = threadIdx.x >> 5;
  int lane = threadIdx.x & 31, half = lane >> 4, ml = lane & 15;
  int stride = gridDim.x * wpb;
  _Float16* st = stage + wib * 16 * 128;
  const int NTILES = Nn / 16;
  for (int base = blockIdx.x * wpb; base < NTILES; base += stride) {
    NO_HOIST();
    int tile = base + wib;
    bool active = tile < NTILES;              // wave-uniform
    float sRow[8]; int gRow[8];
    if (active) {
      int nodeA = tile * 16 + ml;
      const float* xr = x + (size_t)nodeA * 128;
      v16h Ax[4];
#pragma unroll
      for (int c = 0; c < 4; ++c) Ax[c] = frag_a_f32(xr, c * 32, half);
#pragma unroll
      for (int i = 0; i < 8; ++i) {
        int r = tile * 16 + half * 8 + i;
        sRow[i] = S[r]; gRow[i] = batch[r];
      }
#pragma unroll
      for (int nt = 0; nt < 8; ++nt) {
        v8f acc = {};
#pragma unroll
        for (int c = 0; c < 4; ++c) acc = wmma16(Ax[c], Bn[(c * 8 + nt) * 32 + lane], acc);
        int n = nt * 16 + ml;
        float bias = b_n[n];
#pragma unroll
        for (int i = 0; i < 8; ++i)           // transpose C-layout -> row-major in LDS
          st[(half * 8 + i) * 128 + n] = (_Float16)((acc[i] + bias) * sRow[i]);
      }
    }
    __syncthreads();
    if (active) {
      v16h A2[6];
      const _Float16* srow = st + ml * 128;
#pragma unroll
      for (int c = 0; c < 4; ++c) A2[c] = frag_a_f16(srow, c * 32, half);
      {
        int nodeA = tile * 16 + ml;
        const float* gr = glob + (size_t)batch[nodeA] * 64;
        A2[4] = frag_a_f32(gr, 0, half);
        A2[5] = frag_a_f32(gr, 32, half);
      }
#pragma unroll
      for (int nt = 0; nt < 8; ++nt) {
        v8f acc = {};
#pragma unroll
        for (int c = 0; c < 6; ++c) acc = wmma16(A2[c], Bn2[(c * 8 + nt) * 32 + lane], acc);
        int n = nt * 16 + ml;
        float bias = b_n2[n];
#pragma unroll
        for (int i = 0; i < 8; ++i) {
          float v = acc[i] + bias; v = v > 0.f ? v : 0.f;
          out_x[(size_t)(tile * 16 + half * 8 + i) * 128 + n] = v;
          atomicAdd(&gsum[(size_t)gRow[i] * 128 + n], v);
        }
      }
    }
    __syncthreads();                          // protect staging WAR across iterations
  }
}

// ---------- global model: g_new = relu([glob || mean]@W_g + b_g) ----------
__global__ void k_global(const float* __restrict__ glob, const float* __restrict__ gsum,
                         const float* __restrict__ cnt, const float* __restrict__ W_g,
                         const float* __restrict__ b_g, float* __restrict__ out_g) {
  int wib = threadIdx.x >> 5;                 // 4 waves, 16 graph-rows each
  int lane = threadIdx.x & 31, half = lane >> 4, ml = lane & 15;
  int grow = wib * 16 + ml;
  const float* gl = glob + (size_t)grow * 64;
  const float* gs = gsum + (size_t)grow * 128;
  float inv = 1.f / fmaxf(cnt[grow], 1.f);
  v16h A[6];
  A[0] = frag_a_f32(gl, 0, half);
  A[1] = frag_a_f32(gl, 32, half);
#pragma unroll
  for (int c = 0; c < 4; ++c) {               // mean chunks, scaled inline
    v16h a; int off = c * 32 + half * 8;
#pragma unroll
    for (int j = 0; j < 8; ++j) a[j]     = (_Float16)(gs[off + j] * inv);
#pragma unroll
    for (int j = 0; j < 8; ++j) a[j + 8] = (_Float16)(gs[off + 16 + j] * inv);
    A[2 + c] = a;
  }
#pragma unroll
  for (int nt = 0; nt < 4; ++nt) {
    v8f acc = {};
#pragma unroll
    for (int c = 0; c < 6; ++c) acc = wmma16(A[c], frag_b_f32(W_g, 64, c, nt, lane), acc);
    int n = nt * 16 + ml;
    float bias = b_g[n];
#pragma unroll
    for (int i = 0; i < 8; ++i) {
      float v = acc[i] + bias; v = v > 0.f ? v : 0.f;
      out_g[(size_t)(wib * 16 + half * 8 + i) * 64 + n] = v;
    }
  }
}

extern "C" void kernel_launch(void* const* d_in, const int* in_sizes, int n_in,
                              void* d_out, int out_size, void* d_ws, size_t ws_size,
                              hipStream_t stream) {
  const float* x    = (const float*)d_in[0];
  const int*   ei   = (const int*)d_in[1];    // [2, E]
  const float* ea   = (const float*)d_in[2];
  const float* glob = (const float*)d_in[3];
  const int*   batch= (const int*)d_in[4];
  const float* W_e  = (const float*)d_in[5];
  const float* b_e  = (const float*)d_in[6];
  const float* W_le = (const float*)d_in[7];
  const float* b_le = (const float*)d_in[8];
  const float* W_n  = (const float*)d_in[9];
  const float* b_n  = (const float*)d_in[10];
  const float* W_n2 = (const float*)d_in[11];
  const float* b_n2 = (const float*)d_in[12];
  const float* W_g  = (const float*)d_in[13];
  const float* b_g  = (const float*)d_in[14];
  const float* avec = (const float*)d_in[15];

  float* ws    = (float*)d_ws;
  float* smax  = ws;                 // N
  float* denom = ws + Nn;            // N
  float* S     = ws + 2 * Nn;        // N
  float* gsum  = ws + 3 * Nn;        // B*128
  float* cnt   = gsum + Bb * 128;    // B
  float* score = cnt + Bb;           // E
  float* ex    = score + Ee;         // E

  float* out   = (float*)d_out;
  float* out_x = out;                          // N*128
  float* out_e = out + (size_t)Nn * 128;       // E*128
  float* out_g = out_e + (size_t)Ee * 128;     // B*64

  k_init<<<(Nn + 255) / 256, 256, 0, stream>>>(smax, denom, S, gsum, cnt);
  // 625 blocks * 8 waves = 5000 waves -> 10 tiles/wave (50000 tiles), fill amortized 10x
  k_edge1<<<625, 256, 6 * 8 * 32 * sizeof(v16h), stream>>>(x, ei, ea, W_e, b_e, out_e);
  k_edge2<<<625, 256, 12 * 8 * 32 * sizeof(v16h), stream>>>(x, ei, out_e, W_le, b_le,
                                                            avec, score, smax);
  k_exp<<<(Ee + 255) / 256, 256, 0, stream>>>(score, ei, smax, ex, denom);
  k_alpha<<<(Ee + 255) / 256, 256, 0, stream>>>(ei, ex, denom, S);
  k_cnt<<<(Nn + 255) / 256, 256, 0, stream>>>(batch, cnt);
  size_t nodeLds = (4 * 8 * 32 + 6 * 8 * 32) * sizeof(v16h) + 4 * 16 * 128 * 2;
  // 125 blocks * 4 waves = 500 waves over 3125 tiles (~6-7 tiles/wave)
  k_node<<<125, 128, nodeLds, stream>>>(x, batch, glob, W_n, b_n,
                                        W_n2, b_n2, S, out_x, gsum);
  k_global<<<1, 128, 0, stream>>>(glob, gsum, cnt, W_g, b_g, out_g);
}
